// FeaturePropagation_2997887173052
// MI455X (gfx1250) — matrix operand order repaired and checked by reading, not compiled, not run on hardware
//
#include <hip/hip_runtime.h>
#include <hip/hip_bf16.h>

typedef __bf16 bf16;
typedef __attribute__((ext_vector_type(16))) __bf16 v16bf;
typedef __attribute__((ext_vector_type(8)))  float  v8f;
typedef __attribute__((ext_vector_type(4)))  float  f32x4;   // native vector for NT store

// Problem constants (match reference setup_inputs)
#define BB   16
#define N1   1024
#define NP   4096
#define D1   256
#define D2   128
#define HCH  256
#define XW   (D1 + D2)     // 384
#define MROW (BB * NP)     // 65536

// ---------------------------------------------------------------------------
// Weight transpose + f32->bf16 convert:  W (O x Cin) -> Wt (Cin x O) bf16
// ---------------------------------------------------------------------------
__global__ __launch_bounds__(256)
void k_wt_convert(const float* __restrict__ W, bf16* __restrict__ Wt, int O, int Cin) {
    int t = blockIdx.x * 256 + threadIdx.x;
    if (t < O * Cin) {
        int o = t / Cin, c = t % Cin;
        Wt[(size_t)c * O + o] = (bf16)W[t];
    }
}

// ---------------------------------------------------------------------------
// One wave per query point: 3-NN inverse-distance top-k + interpolation +
// concat with features2, emitting bf16 X (MROW x 384).
// xyz1 for the block's batch is staged in LDS (all 8 waves share one batch).
// ---------------------------------------------------------------------------
__global__ __launch_bounds__(256)
void k_topk_interp(const float* __restrict__ xyz1, const float* __restrict__ xyz2,
                   const float* __restrict__ f1,   const float* __restrict__ f2,
                   bf16* __restrict__ X) {
    __shared__ float sx[N1 * 3];                     // 12 KB
    int tid   = threadIdx.x;
    int gwave = (blockIdx.x * 256 + tid) >> 5;       // 0..65535
    int lane  = tid & 31;
    int b = gwave >> 12;                             // same for whole block
    int j = gwave & (NP - 1);

    const float* p = xyz1 + (size_t)b * N1 * 3;
    for (int idx = tid; idx < N1 * 3; idx += 256) sx[idx] = p[idx];   // coalesced
    __syncthreads();

    const float* q = xyz2 + ((size_t)b * NP + j) * 3;
    float qx = q[0], qy = q[1], qz = q[2];

    float v0 = -1.f, v1 = -1.f, v2 = -1.f;
    int   i0 = 0,    i1 = 0,    i2 = 0;

    auto ins = [&](float u, int o) {
        if (u > v0)      { v2 = v1; i2 = i1; v1 = v0; i1 = i0; v0 = u; i0 = o; }
        else if (u > v1) { v2 = v1; i2 = i1; v1 = u;  i1 = o; }
        else if (u > v2) { v2 = u;  i2 = o; }
    };

    for (int i = lane; i < N1; i += 32) {            // bank stride 3: conflict-free
        float dx = qx - sx[i * 3 + 0];
        float dy = qy - sx[i * 3 + 1];
        float dz = qz - sx[i * 3 + 2];
        float inv = 1.0f / (dx * dx + dy * dy + dz * dz + 1.1920929e-07f);
        ins(inv, i);
    }
    // Butterfly merge of per-lane sorted top-3 lists (wave32)
    #pragma unroll
    for (int m = 16; m >= 1; m >>= 1) {
        float u0 = __shfl_xor(v0, m, 32);
        float u1 = __shfl_xor(v1, m, 32);
        float u2 = __shfl_xor(v2, m, 32);
        int   o0 = __shfl_xor(i0, m, 32);
        int   o1 = __shfl_xor(i1, m, 32);
        int   o2 = __shfl_xor(i2, m, 32);
        ins(u0, o0); ins(u1, o1); ins(u2, o2);
    }
    float s  = v0 + v1 + v2;
    float w0 = v0 / s, w1 = v1 / s, w2 = v2 / s;

    const float4* r0 = (const float4*)(f1 + ((size_t)b * N1 + i0) * D1);
    const float4* r1 = (const float4*)(f1 + ((size_t)b * N1 + i1) * D1);
    const float4* r2 = (const float4*)(f1 + ((size_t)b * N1 + i2) * D1);
    bf16* xo = X + ((size_t)b * NP + j) * XW;

    #pragma unroll
    for (int u = 0; u < 2; ++u) {                    // 8 interp channels / lane
        int c4 = lane * 2 + u;
        float4 a = r0[c4], bq = r1[c4], cq = r2[c4];
        int c = c4 * 4;
        xo[c + 0] = (bf16)(w0 * a.x + w1 * bq.x + w2 * cq.x);
        xo[c + 1] = (bf16)(w0 * a.y + w1 * bq.y + w2 * cq.y);
        xo[c + 2] = (bf16)(w0 * a.z + w1 * bq.z + w2 * cq.z);
        xo[c + 3] = (bf16)(w0 * a.w + w1 * bq.w + w2 * cq.w);
    }
    float4 f2v = ((const float4*)(f2 + ((size_t)b * NP + j) * D2))[lane];
    int c = D1 + lane * 4;                           // 128 passthrough channels
    xo[c + 0] = (bf16)f2v.x;
    xo[c + 1] = (bf16)f2v.y;
    xo[c + 2] = (bf16)f2v.z;
    xo[c + 3] = (bf16)f2v.w;
}

// ---------------------------------------------------------------------------
// WMMA GEMM: C(f32, M x 256) = A(bf16, M x K, row-major) * Bt(bf16, K x 256)
// Block: 256 threads (8 waves), tile 128x128, K-step 32.
// Wave (wm,wn): rows wm*32..+31, cols wn*64..+63 -> 8 WMMA tiles per K-step.
// A tile copied global->LDS with GLOBAL_LOAD_ASYNC_TO_LDS_B128 (ASYNCcnt);
// B tile transposed in registers (pair-pack) -> aligned b32 LDS stores.
// N hardcoded to 256 so the epilogue is base + immediate-offset stores.
// ---------------------------------------------------------------------------
#define BM 128
#define BN 128
#define BK 32
#define LDK 40            // padded LDS row stride in bf16 (80 B: keeps b128 align)

__global__ __launch_bounds__(256)
void k_wmma_gemm(const bf16* __restrict__ A, const bf16* __restrict__ Bt,
                 float* __restrict__ C, int K) {
    __shared__ bf16 sA[BM * LDK];     // [m][k]   10240 B
    __shared__ bf16 sB[BN * LDK];     // [n][k]   10240 B (K-pairs contiguous)

    int tid  = threadIdx.x;
    int wave = tid >> 5;
    int lane = tid & 31;
    int bm = blockIdx.x * BM;
    int bn = blockIdx.y * BN;
    int wm = wave >> 1;               // 0..3  (32-row strip)
    int wn = wave & 1;                // 0..1  (64-col strip)
    int half = lane >> 4;
    int l15  = lane & 15;

    // A coop async load: 16 bf16 (32 B) per thread, contiguous
    int arow = tid >> 1, acol = (tid & 1) * 16;
    unsigned ldsA = (unsigned)(uintptr_t)&sA[arow * LDK + acol];   // low 32 bits = DS offset
    unsigned long long gA = (unsigned long long)(A + (size_t)(bm + arow) * K + acol);

    // B coop load: 2(k) x 8(n) micro-tile per thread
    int kp = tid >> 4;                // 0..15 -> k = 2*kp
    int ng = tid & 15;                // 0..15 -> n0 = 8*ng
    int brow = kp * 2, bn0 = ng * 8;
    const bf16* gB = Bt + (size_t)brow * HCH + bn + bn0;
    unsigned* sBw = (unsigned*)sB;

    v8f acc[2][4];
    #pragma unroll
    for (int mt = 0; mt < 2; ++mt)
        #pragma unroll
        for (int nt = 0; nt < 4; ++nt)
            acc[mt][nt] = (v8f){0.f, 0.f, 0.f, 0.f, 0.f, 0.f, 0.f, 0.f};

    for (int k0 = 0; k0 < K; k0 += BK) {
        // ---- A tile: async global->LDS (2 x b128), tracked on ASYNCcnt ----
        asm volatile(
            "global_load_async_to_lds_b128 %0, %1, off\n\t"
            "global_load_async_to_lds_b128 %0, %1, off offset:16"
            :: "v"(ldsA), "v"(gA) : "memory");

        // ---- B tile: 2 rows x 8 cols, pack (k,k+1) pairs, aligned b32 stores ----
        uint4 q0 = *(const uint4*)(gB);
        uint4 q1 = *(const uint4*)(gB + HCH);
        unsigned rw0[4] = {q0.x, q0.y, q0.z, q0.w};
        unsigned rw1[4] = {q1.x, q1.y, q1.z, q1.w};
        #pragma unroll
        for (int i = 0; i < 8; ++i) {
            unsigned lo = (rw0[i >> 1] >> (16 * (i & 1))) & 0xffffu;
            unsigned hi = (rw1[i >> 1] >> (16 * (i & 1))) & 0xffffu;
            sBw[((bn0 + i) * LDK + brow) >> 1] = lo | (hi << 16);
        }

        asm volatile("s_wait_asynccnt 0x0" ::: "memory");
        __syncthreads();

        // ---- A fragments (16x32 bf16 per ISA layout), 2 per wave ----
        v16bf af[2];
        #pragma unroll
        for (int mt = 0; mt < 2; ++mt) {
            int mloc = wm * 32 + mt * 16 + l15;
            #pragma unroll
            for (int jj = 0; jj < 8; ++jj) {
                int kb = ((jj < 4) ? (jj * 2) : (16 + (jj - 4) * 2)) + half * 8;
                af[mt][2 * jj]     = sA[mloc * LDK + kb];
                af[mt][2 * jj + 1] = sA[mloc * LDK + kb + 1];
            }
        }
        // ---- B fragments + 8 WMMAs ----
        #pragma unroll
        for (int nt = 0; nt < 4; ++nt) {
            int nloc = wn * 64 + nt * 16 + l15;
            v16bf bf;
            #pragma unroll
            for (int jj = 0; jj < 8; ++jj) {
                int kk = ((jj < 4) ? (jj * 2) : (16 + (jj - 4) * 2)) + half * 8;
                bf[2 * jj]     = sB[nloc * LDK + kk];
                bf[2 * jj + 1] = sB[nloc * LDK + kk + 1];
            }
            acc[0][nt] = __builtin_amdgcn_wmma_f32_16x16x32_bf16(
                false, af[0], false, bf, (short)0, acc[0][nt], false, false);
            acc[1][nt] = __builtin_amdgcn_wmma_f32_16x16x32_bf16(
                false, af[1], false, bf, (short)0, acc[1][nt], false, false);
        }
        __syncthreads();

        gA += (unsigned long long)BK * 2;     // bytes
        gB += (size_t)BK * HCH;
    }

    // C/D 16x16 f32 layout: lane n = lane&15, VGPR v -> row v + 8*(lane>>4).
    // One base address per mt; all stores are immediate offsets (N=256 const).
    #pragma unroll
    for (int mt = 0; mt < 2; ++mt) {
        float* cp = C + (size_t)(bm + wm * 32 + mt * 16 + half * 8) * HCH
                      + bn + wn * 64 + l15;
        #pragma unroll
        for (int nt = 0; nt < 4; ++nt)
            #pragma unroll
            for (int v = 0; v < 8; ++v)
                cp[v * HCH + nt * 16] = acc[mt][nt][v];
    }
}

// ---------------------------------------------------------------------------
// BatchNorm: zero accumulators, float4 reduction with LDS partial combine,
// fused finalize+normalize+ReLU apply.
// ---------------------------------------------------------------------------
__global__ __launch_bounds__(256)
void k_bn_init(float* __restrict__ accum) {
    accum[blockIdx.x * 256 + threadIdx.x] = 0.f;
}

// Block: 256 rows x 256 channels. Thread -> 4 channels, 64 rows (float4 loads).
__global__ __launch_bounds__(256)
void k_bn_stats(const float* __restrict__ Y, float* __restrict__ accum) {
    __shared__ float red[4][64][8];       // [rsub][cgroup][s0..3, q0..3]
    int t    = threadIdx.x;
    int cg   = t & 63;                    // channel group -> channels cg*4..+3
    int rsub = t >> 6;                    // 0..3
    int c4   = cg * 4;
    size_t row0 = (size_t)blockIdx.x * 256 + rsub;

    float s0 = 0.f, s1 = 0.f, s2 = 0.f, s3 = 0.f;
    float q0 = 0.f, q1 = 0.f, q2 = 0.f, q3 = 0.f;
    for (int r = 0; r < 64; ++r) {
        float4 y = *(const float4*)(Y + (row0 + (size_t)r * 4) * HCH + c4);
        s0 += y.x; q0 += y.x * y.x;
        s1 += y.y; q1 += y.y * y.y;
        s2 += y.z; q2 += y.z * y.z;
        s3 += y.w; q3 += y.w * y.w;
    }
    red[rsub][cg][0] = s0; red[rsub][cg][1] = s1;
    red[rsub][cg][2] = s2; red[rsub][cg][3] = s3;
    red[rsub][cg][4] = q0; red[rsub][cg][5] = q1;
    red[rsub][cg][6] = q2; red[rsub][cg][7] = q3;
    __syncthreads();
    if (t < 64) {
        #pragma unroll
        for (int e = 0; e < 4; ++e) {
            float s = red[0][t][e]     + red[1][t][e]     + red[2][t][e]     + red[3][t][e];
            float q = red[0][t][e + 4] + red[1][t][e + 4] + red[2][t][e + 4] + red[3][t][e + 4];
            atomicAdd(&accum[t * 4 + e], s);
            atomicAdd(&accum[HCH + t * 4 + e], q);
        }
    }
}

// Block: 64 rows x 256 channels. Thread -> 4 channels, 16 rows; stats folded
// once into y' = y*a + d form.
__global__ __launch_bounds__(256)
void k_bn_apply_bf16(const float* __restrict__ Y, const float* __restrict__ accum,
                     const float* __restrict__ g, const float* __restrict__ beta,
                     bf16* __restrict__ Xout) {
    int t    = threadIdx.x;
    int c4   = (t & 63) * 4;
    int rsub = t >> 6;
    const float invM = 1.0f / (float)MROW;
    float4 sm = *(const float4*)(accum + c4);
    float4 sq = *(const float4*)(accum + HCH + c4);
    float4 gg = *(const float4*)(g + c4);
    float4 bb = *(const float4*)(beta + c4);
    float mu0 = sm.x * invM, mu1 = sm.y * invM, mu2 = sm.z * invM, mu3 = sm.w * invM;
    float a0 = rsqrtf(sq.x * invM - mu0 * mu0 + 1e-5f) * gg.x;
    float a1 = rsqrtf(sq.y * invM - mu1 * mu1 + 1e-5f) * gg.y;
    float a2 = rsqrtf(sq.z * invM - mu2 * mu2 + 1e-5f) * gg.z;
    float a3 = rsqrtf(sq.w * invM - mu3 * mu3 + 1e-5f) * gg.w;
    float d0 = bb.x - mu0 * a0, d1 = bb.y - mu1 * a1;
    float d2 = bb.z - mu2 * a2, d3 = bb.w - mu3 * a3;

    for (int r = 0; r < 16; ++r) {
        size_t row = (size_t)blockIdx.x * 64 + rsub + r * 4;
        float4 y = *(const float4*)(Y + row * HCH + c4);
        float o0 = y.x * a0 + d0, o1 = y.y * a1 + d1;
        float o2 = y.z * a2 + d2, o3 = y.w * a3 + d3;
        bf16* xo = Xout + row * HCH + c4;
        xo[0] = (bf16)(o0 > 0.f ? o0 : 0.f);
        xo[1] = (bf16)(o1 > 0.f ? o1 : 0.f);
        xo[2] = (bf16)(o2 > 0.f ? o2 : 0.f);
        xo[3] = (bf16)(o3 > 0.f ? o3 : 0.f);
    }
}

__global__ __launch_bounds__(256)
void k_bn_apply_f32(float* __restrict__ Y, const float* __restrict__ accum,
                    const float* __restrict__ g, const float* __restrict__ beta) {
    int t    = threadIdx.x;
    int c4   = (t & 63) * 4;
    int rsub = t >> 6;
    const float invM = 1.0f / (float)MROW;
    float4 sm = *(const float4*)(accum + c4);
    float4 sq = *(const float4*)(accum + HCH + c4);
    float4 gg = *(const float4*)(g + c4);
    float4 bb = *(const float4*)(beta + c4);
    float mu0 = sm.x * invM, mu1 = sm.y * invM, mu2 = sm.z * invM, mu3 = sm.w * invM;
    float a0 = rsqrtf(sq.x * invM - mu0 * mu0 + 1e-5f) * gg.x;
    float a1 = rsqrtf(sq.y * invM - mu1 * mu1 + 1e-5f) * gg.y;
    float a2 = rsqrtf(sq.z * invM - mu2 * mu2 + 1e-5f) * gg.z;
    float a3 = rsqrtf(sq.w * invM - mu3 * mu3 + 1e-5f) * gg.w;
    float d0 = bb.x - mu0 * a0, d1 = bb.y - mu1 * a1;
    float d2 = bb.z - mu2 * a2, d3 = bb.w - mu3 * a3;

    for (int r = 0; r < 16; ++r) {
        size_t row = (size_t)blockIdx.x * 64 + rsub + r * 4;
        float4 y = *(const float4*)(Y + row * HCH + c4);
        f32x4 o;
        o.x = y.x * a0 + d0; o.x = o.x > 0.f ? o.x : 0.f;
        o.y = y.y * a1 + d1; o.y = o.y > 0.f ? o.y : 0.f;
        o.z = y.z * a2 + d2; o.z = o.z > 0.f ? o.z : 0.f;
        o.w = y.w * a3 + d3; o.w = o.w > 0.f ? o.w : 0.f;
        // final output is never re-read: bypass cache with NT store
        __builtin_nontemporal_store(o, (f32x4*)(Y + row * HCH + c4));
    }
}

// ---------------------------------------------------------------------------
// Launcher
// ---------------------------------------------------------------------------
extern "C" void kernel_launch(void* const* d_in, const int* in_sizes, int n_in,
                              void* d_out, int out_size, void* d_ws, size_t ws_size,
                              hipStream_t stream) {
    (void)in_sizes; (void)n_in; (void)out_size; (void)ws_size;

    const float* xyz1  = (const float*)d_in[0];
    const float* xyz2  = (const float*)d_in[1];
    const float* feat1 = (const float*)d_in[2];
    const float* feat2 = (const float*)d_in[3];
    const float* W1    = (const float*)d_in[4];
    // b1 = d_in[5], b2 = d_in[9]: bias cancels exactly under BatchNorm -> unused
    const float* g1    = (const float*)d_in[6];
    const float* be1   = (const float*)d_in[7];
    const float* W2    = (const float*)d_in[8];
    const float* g2    = (const float*)d_in[10];
    const float* be2   = (const float*)d_in[11];
    float* out = (float*)d_out;

    char* ws = (char*)d_ws;
    const size_t offX   = 0;                                  // X bf16 65536x384, reused for X2
    const size_t offY1  = offX  + (size_t)MROW * XW * 2;      // Y1 f32 65536x256
    const size_t offW1t = offY1 + (size_t)MROW * HCH * 4;     // W1t bf16 384x256
    const size_t offW2t = offW1t + (size_t)XW * HCH * 2;      // W2t bf16 256x256
    const size_t offAcc = offW2t + (size_t)HCH * HCH * 2;     // 512 f32 accumulators

    bf16*  X     = (bf16*)(ws + offX);
    float* Y1    = (float*)(ws + offY1);
    bf16*  W1t   = (bf16*)(ws + offW1t);
    bf16*  W2t   = (bf16*)(ws + offW2t);
    float* accum = (float*)(ws + offAcc);
    bf16*  X2    = (bf16*)(ws + offX);                        // reuse X region

    // 1) weight convert/transpose
    k_wt_convert<<<(HCH * XW  + 255) / 256, 256, 0, stream>>>(W1, W1t, HCH, XW);
    k_wt_convert<<<(HCH * HCH + 255) / 256, 256, 0, stream>>>(W2, W2t, HCH, HCH);

    // 2) 3-NN top-k + interpolation + concat -> X bf16
    k_topk_interp<<<(MROW * 32) / 256, 256, 0, stream>>>(xyz1, xyz2, feat1, feat2, X);

    // 3) GEMM1: Y1 = X(65536x384) * W1t(384x256)
    k_wmma_gemm<<<dim3(MROW / BM, HCH / BN), 256, 0, stream>>>(X, W1t, Y1, XW);

    // 4) BN1 stats + apply (relu, bf16 out) -> X2
    k_bn_init <<<2, 256, 0, stream>>>(accum);
    k_bn_stats<<<MROW / 256, 256, 0, stream>>>(Y1, accum);
    k_bn_apply_bf16<<<MROW / 64, 256, 0, stream>>>(Y1, accum, g1, be1, X2);

    // 5) GEMM2: out = X2(65536x256) * W2t(256x256)  (pre-BN f32 into d_out)
    k_wmma_gemm<<<dim3(MROW / BM, HCH / BN), 256, 0, stream>>>(X2, W2t, out, HCH);

    // 6) BN2 stats + apply (relu, f32 in-place on d_out)
    k_bn_init <<<2, 256, 0, stream>>>(accum);
    k_bn_stats<<<MROW / 256, 256, 0, stream>>>(out, accum);
    k_bn_apply_f32<<<MROW / 64, 256, 0, stream>>>(out, accum, g2, be2);
}
